// MultiHeadAttention_61237643707076
// MI455X (gfx1250) — matrix operand order repaired and checked by reading, not compiled, not run on hardware
//
#include <hip/hip_runtime.h>
#include <hip/hip_bf16.h>

#define DIM   1024
#define NH    16
#define HD    64
#define SEQ   2048
#define BATCH 4
#define MTOT  (BATCH * SEQ)   // 8192
#define NQKV  (3 * DIM)       // 3072
#define ATTN_SCALE 0.125f     // 1/sqrt(64)

typedef __attribute__((ext_vector_type(16))) __bf16        v16bf;
typedef __attribute__((ext_vector_type(8)))  float         v8f;
typedef __attribute__((ext_vector_type(4)))  unsigned int  u32x4;
typedef __attribute__((ext_vector_type(4)))  unsigned short u16x4;
typedef __attribute__((ext_vector_type(4)))  float         f32x4;

union Frag { v16bf v; u32x4 q[2]; };

__device__ __forceinline__ unsigned short f2bf(float f) {
  unsigned u = __float_as_uint(f);
  u += 0x7FFFu + ((u >> 16) & 1u);   // round-to-nearest-even
  return (unsigned short)(u >> 16);
}

__device__ __forceinline__ v8f wmma_bf16(v16bf a, v16bf b, v8f c) {
  // (neg_a, A, neg_b, B, c_mod, C, reuse_a, reuse_b)
  return __builtin_amdgcn_wmma_f32_16x16x32_bf16(false, a, false, b,
                                                 (short)0, c, false, false);
}

// CDNA5 async global->LDS copy (ASYNCcnt-tracked). vdst = LDS byte offset VGPR.
__device__ __forceinline__ void async_ld_b128(const unsigned short* lds_dst,
                                              const void* gptr) {
  unsigned off = (unsigned)(size_t)lds_dst;   // low 32 bits of generic ptr = LDS offset
  asm volatile("global_load_async_to_lds_b128 %0, %1, off"
               :: "v"(off), "v"(gptr) : "memory");
}
__device__ __forceinline__ void wait_async0() {
  asm volatile("s_wait_asynccnt 0x0" ::: "memory");
}
__device__ __forceinline__ void wait_ds0() {
  asm volatile("s_wait_dscnt 0x0" ::: "memory");
}

// ---------------------------------------------------------------------------
// Kernel 1: qkv = x @ w_qkv^T  (fp32 in -> bf16 staged -> WMMA bf16 -> bf16 out)
// Q,K stored [B,H,N,Dh]; V stored transposed [B,H,Dh,N].
// Workgroup: 256 thr = 8 waves; tile 128(M) x 256(N); wave tile 64x64.
// ---------------------------------------------------------------------------
__global__ __launch_bounds__(256) void qkv_gemm_k(
    const float* __restrict__ x, const float* __restrict__ wqkv,
    unsigned short* __restrict__ qw, unsigned short* __restrict__ kw,
    unsigned short* __restrict__ vtw)
{
  __shared__ unsigned short sA[128 * 40];   // 32 K + 8 pad per row
  __shared__ unsigned short sB[256 * 40];
  const int tid = threadIdx.x;
  const int lane = tid & 31, wave = tid >> 5;
  const int wm = wave >> 2, wn = wave & 3;
  const int m0 = blockIdx.x * 128;
  const int n0 = blockIdx.y * 256;
  const int lhalf = lane >> 4, l16 = lane & 15;

  v8f acc[4][4];
#pragma unroll
  for (int i = 0; i < 4; ++i)
#pragma unroll
    for (int j = 0; j < 4; ++j)
      acc[i][j] = (v8f){0.f, 0.f, 0.f, 0.f, 0.f, 0.f, 0.f, 0.f};

  for (int k0 = 0; k0 < DIM; k0 += 32) {
    // stage A tile 128x32 fp32 -> bf16
#pragma unroll
    for (int i = 0; i < 4; ++i) {
      int idx = tid + i * 256;                 // 1024 chunks of 4 floats
      int row = idx >> 3, ch = idx & 7;
      f32x4 f = *(const f32x4*)(x + (size_t)(m0 + row) * DIM + k0 + ch * 4);
      u16x4 hb = { f2bf(f.x), f2bf(f.y), f2bf(f.z), f2bf(f.w) };
      *(u16x4*)&sA[row * 40 + ch * 4] = hb;
    }
    // stage B tile 256x32 fp32 -> bf16
#pragma unroll
    for (int i = 0; i < 8; ++i) {
      int idx = tid + i * 256;
      int row = idx >> 3, ch = idx & 7;
      f32x4 f = *(const f32x4*)(wqkv + (size_t)(n0 + row) * DIM + k0 + ch * 4);
      u16x4 hb = { f2bf(f.x), f2bf(f.y), f2bf(f.z), f2bf(f.w) };
      *(u16x4*)&sB[row * 40 + ch * 4] = hb;
    }
    if (k0 + 32 < DIM)
      __builtin_prefetch(x + (size_t)(m0 + (tid >> 3)) * DIM + k0 + 32, 0, 1);
    __syncthreads();

    Frag aF[4];
#pragma unroll
    for (int ms = 0; ms < 4; ++ms) {
      int r = wm * 64 + ms * 16 + l16;
      int kb = lhalf * 8;
      aF[ms].q[0] = *(const u32x4*)&sA[r * 40 + kb];
      aF[ms].q[1] = *(const u32x4*)&sA[r * 40 + kb + 16];
    }
#pragma unroll
    for (int ns = 0; ns < 4; ++ns) {
      Frag bF;
      int r = wn * 64 + ns * 16 + l16;
      int kk = lhalf * 8;
      bF.q[0] = *(const u32x4*)&sB[r * 40 + kk];
      bF.q[1] = *(const u32x4*)&sB[r * 40 + kk + 16];
#pragma unroll
      for (int ms = 0; ms < 4; ++ms)
        acc[ms][ns] = wmma_bf16(aF[ms].v, bF.v, acc[ms][ns]);
    }
    __syncthreads();
  }

  const int sec = n0 >> 10;   // 0=q 1=k 2=v (uniform per block: 256 | 1024)
#pragma unroll
  for (int ms = 0; ms < 4; ++ms)
#pragma unroll
    for (int ns = 0; ns < 4; ++ns)
#pragma unroll
      for (int e = 0; e < 8; ++e) {
        int m = m0 + wm * 64 + ms * 16 + e + 8 * lhalf;
        int c = (n0 & 1023) + wn * 64 + ns * 16 + l16;
        int b = m >> 11, t = m & 2047;
        int h = c >> 6, d = c & 63;
        unsigned short val = f2bf(acc[ms][ns][e]);
        if (sec == 0)      qw[((size_t)(b * NH + h) * SEQ + t) * HD + d] = val;
        else if (sec == 1) kw[((size_t)(b * NH + h) * SEQ + t) * HD + d] = val;
        else               vtw[((size_t)(b * NH + h) * HD + d) * SEQ + t] = val;
      }
}

// ---------------------------------------------------------------------------
// Kernel 2: flash attention per (b,h). Block = 128 q rows (8 waves x 16 rows).
// K/V staged via async global->LDS; online softmax; all GEMMs via WMMA bf16.
// ---------------------------------------------------------------------------
__global__ __launch_bounds__(256) void attn_k(
    const unsigned short* __restrict__ qw, const unsigned short* __restrict__ kw,
    const unsigned short* __restrict__ vtw, unsigned short* __restrict__ ow)
{
  __shared__ unsigned short sK[32 * 72];       // 32 keys x (64 dh + 8 pad)
  __shared__ unsigned short sV[64 * 40];       // 64 dh x (32 keys + 8 pad)
  __shared__ unsigned short sP[8 * 16 * 40];   // per-wave P relayout buffer
  const int tid = threadIdx.x;
  const int lane = tid & 31, wave = tid >> 5;
  const int lhalf = lane >> 4, l16 = lane & 15;
  const int bh = blockIdx.y;
  const int b = bh >> 4, h = bh & 15;
  const int q0 = blockIdx.x * 128 + wave * 16;

  const unsigned short* Qb = qw + (size_t)bh * SEQ * HD;
  const unsigned short* Kb = kw + (size_t)bh * SEQ * HD;
  const unsigned short* Vb = vtw + (size_t)bh * HD * SEQ;
  unsigned short* myP = &sP[wave * 16 * 40];

  // Q fragments (A-matrix, Dh=64 -> 2 k-steps of 32)
  Frag qf[2];
  {
    int qr = q0 + l16;
    int kb = lhalf * 8;
    qf[0].q[0] = *(const u32x4*)(Qb + (size_t)qr * HD + kb);
    qf[0].q[1] = *(const u32x4*)(Qb + (size_t)qr * HD + kb + 16);
    qf[1].q[0] = *(const u32x4*)(Qb + (size_t)qr * HD + 32 + kb);
    qf[1].q[1] = *(const u32x4*)(Qb + (size_t)qr * HD + 32 + kb + 16);
  }

  v8f oacc[4];
#pragma unroll
  for (int d = 0; d < 4; ++d)
    oacc[d] = (v8f){0.f, 0.f, 0.f, 0.f, 0.f, 0.f, 0.f, 0.f};
  float rmax[8], rsum[8];
#pragma unroll
  for (int e = 0; e < 8; ++e) { rmax[e] = -1e30f; rsum[e] = 0.f; }

  for (int kb0 = 0; kb0 < SEQ; kb0 += 32) {
    {   // cooperative async staging of K (32x64) and V^T (64x32)
      int key = tid >> 3, c = tid & 7;
      async_ld_b128(&sK[key * 72 + c * 8],
                    Kb + (size_t)(kb0 + key) * HD + c * 8);
      int dh = tid >> 2, c2 = tid & 3;
      async_ld_b128(&sV[dh * 40 + c2 * 8],
                    Vb + (size_t)dh * SEQ + kb0 + c2 * 8);
    }
    wait_async0();
    __syncthreads();

    // S = (Q K^T) * scale for 2 tiles of 16 keys
    v8f sv[2];
#pragma unroll
    for (int t2 = 0; t2 < 2; ++t2) {
      Frag kf0, kf1;
      int kl = t2 * 16 + l16;
      int kk = lhalf * 8;
      kf0.q[0] = *(const u32x4*)&sK[kl * 72 + kk];
      kf0.q[1] = *(const u32x4*)&sK[kl * 72 + kk + 16];
      kf1.q[0] = *(const u32x4*)&sK[kl * 72 + 32 + kk];
      kf1.q[1] = *(const u32x4*)&sK[kl * 72 + 32 + kk + 16];
      v8f z = (v8f){0.f, 0.f, 0.f, 0.f, 0.f, 0.f, 0.f, 0.f};
      z = wmma_bf16(qf[0].v, kf0.v, z);
      z = wmma_bf16(qf[1].v, kf1.v, z);
#pragma unroll
      for (int e = 0; e < 8; ++e) z[e] *= ATTN_SCALE;
      sv[t2] = z;
    }

    // online softmax: row max over 16 lanes of each half-wave
    float bm[8];
#pragma unroll
    for (int e = 0; e < 8; ++e) bm[e] = fmaxf(sv[0][e], sv[1][e]);
#pragma unroll
    for (int msk = 1; msk <= 8; msk <<= 1)
#pragma unroll
      for (int e = 0; e < 8; ++e)
        bm[e] = fmaxf(bm[e], __shfl_xor(bm[e], msk, 16));

    float fac[8], psum[8];
#pragma unroll
    for (int e = 0; e < 8; ++e) {
      float mn = fmaxf(rmax[e], bm[e]);
      fac[e] = __expf(rmax[e] - mn);
      rmax[e] = mn;
      float p0 = __expf(sv[0][e] - mn);
      float p1 = __expf(sv[1][e] - mn);
      int row = e + 8 * lhalf;                   // C-layout row
      myP[row * 40 + l16]      = f2bf(p0);       // key l16
      myP[row * 40 + 16 + l16] = f2bf(p1);       // key 16+l16
      psum[e] = p0 + p1;
    }
#pragma unroll
    for (int msk = 1; msk <= 8; msk <<= 1)
#pragma unroll
      for (int e = 0; e < 8; ++e) psum[e] += __shfl_xor(psum[e], msk, 16);
#pragma unroll
    for (int e = 0; e < 8; ++e) rsum[e] = rsum[e] * fac[e] + psum[e];
#pragma unroll
    for (int d = 0; d < 4; ++d)
#pragma unroll
      for (int e = 0; e < 8; ++e) oacc[d][e] *= fac[e];

    wait_ds0();   // per-wave LDS RAW fence for P relayout

    Frag pf;      // P as A-fragment (16 x 32 keys)
    {
      int kb = lhalf * 8;
      pf.q[0] = *(const u32x4*)&myP[l16 * 40 + kb];
      pf.q[1] = *(const u32x4*)&myP[l16 * 40 + kb + 16];
    }
#pragma unroll
    for (int d = 0; d < 4; ++d) {   // O += P * V over 4 dh-subtiles
      Frag vf;
      int dh = d * 16 + l16;
      int kk = lhalf * 8;
      vf.q[0] = *(const u32x4*)&sV[dh * 40 + kk];
      vf.q[1] = *(const u32x4*)&sV[dh * 40 + kk + 16];
      oacc[d] = wmma_bf16(pf.v, vf.v, oacc[d]);
    }
    __syncthreads();   // protect sK/sV before next stage overwrites
  }

#pragma unroll
  for (int e = 0; e < 8; ++e) rsum[e] = 1.f / rsum[e];
#pragma unroll
  for (int d = 0; d < 4; ++d)
#pragma unroll
    for (int e = 0; e < 8; ++e) {
      int row = q0 + e + 8 * lhalf;
      int dh = d * 16 + l16;
      ow[((size_t)(b * SEQ) + row) * DIM + h * HD + dh] =
          f2bf(oacc[d][e] * rsum[e]);
    }
}

// ---------------------------------------------------------------------------
// Kernel 3: out = O @ w_proj^T + b_proj  (A is bf16 -> async-to-LDS staging)
// ---------------------------------------------------------------------------
__global__ __launch_bounds__(256) void proj_k(
    const unsigned short* __restrict__ ow, const float* __restrict__ wp,
    const float* __restrict__ bp, float* __restrict__ out)
{
  __shared__ unsigned short sA[128 * 40];
  __shared__ unsigned short sB[256 * 40];
  const int tid = threadIdx.x;
  const int lane = tid & 31, wave = tid >> 5;
  const int wm = wave >> 2, wn = wave & 3;
  const int m0 = blockIdx.x * 128;
  const int n0 = blockIdx.y * 256;
  const int lhalf = lane >> 4, l16 = lane & 15;

  v8f acc[4][4];
#pragma unroll
  for (int i = 0; i < 4; ++i)
#pragma unroll
    for (int j = 0; j < 4; ++j)
      acc[i][j] = (v8f){0.f, 0.f, 0.f, 0.f, 0.f, 0.f, 0.f, 0.f};

  for (int k0 = 0; k0 < DIM; k0 += 32) {
#pragma unroll
    for (int i = 0; i < 2; ++i) {     // A tile 128x32 bf16, async
      int idx = tid + i * 256;
      int row = idx >> 2, c = idx & 3;
      async_ld_b128(&sA[row * 40 + c * 8],
                    ow + (size_t)(m0 + row) * DIM + k0 + c * 8);
    }
#pragma unroll
    for (int i = 0; i < 8; ++i) {     // B tile 256x32 fp32 -> bf16
      int idx = tid + i * 256;
      int row = idx >> 3, ch = idx & 7;
      f32x4 f = *(const f32x4*)(wp + (size_t)(n0 + row) * DIM + k0 + ch * 4);
      u16x4 hb = { f2bf(f.x), f2bf(f.y), f2bf(f.z), f2bf(f.w) };
      *(u16x4*)&sB[row * 40 + ch * 4] = hb;
    }
    wait_async0();
    __syncthreads();

    Frag aF[4];
#pragma unroll
    for (int ms = 0; ms < 4; ++ms) {
      int r = wm * 64 + ms * 16 + l16;
      int kb = lhalf * 8;
      aF[ms].q[0] = *(const u32x4*)&sA[r * 40 + kb];
      aF[ms].q[1] = *(const u32x4*)&sA[r * 40 + kb + 16];
    }
#pragma unroll
    for (int ns = 0; ns < 4; ++ns) {
      Frag bF;
      int r = wn * 64 + ns * 16 + l16;
      int kk = lhalf * 8;
      bF.q[0] = *(const u32x4*)&sB[r * 40 + kk];
      bF.q[1] = *(const u32x4*)&sB[r * 40 + kk + 16];
#pragma unroll
      for (int ms = 0; ms < 4; ++ms)
        acc[ms][ns] = wmma_bf16(aF[ms].v, bF.v, acc[ms][ns]);
    }
    __syncthreads();
  }

#pragma unroll
  for (int ms = 0; ms < 4; ++ms)
#pragma unroll
    for (int ns = 0; ns < 4; ++ns) {
      int n = n0 + wn * 64 + ns * 16 + l16;
      float bias = bp[n];
#pragma unroll
      for (int e = 0; e < 8; ++e) {
        int m = m0 + wm * 64 + ms * 16 + e + 8 * lhalf;
        out[(size_t)m * DIM + n] = acc[ms][ns][e] + bias;
      }
    }
}

// ---------------------------------------------------------------------------
extern "C" void kernel_launch(void* const* d_in, const int* in_sizes, int n_in,
                              void* d_out, int out_size, void* d_ws, size_t ws_size,
                              hipStream_t stream) {
  (void)in_sizes; (void)n_in; (void)out_size; (void)ws_size;
  const float* x    = (const float*)d_in[0];
  const float* wqkv = (const float*)d_in[1];
  const float* wp   = (const float*)d_in[2];
  const float* bp   = (const float*)d_in[3];
  float* out = (float*)d_out;

  const size_t nel = (size_t)BATCH * NH * SEQ * HD;   // 8,388,608
  char* ws = (char*)d_ws;
  unsigned short* qw  = (unsigned short*)ws;  ws += nel * 2;   // Q  [B,H,N,Dh]
  unsigned short* kw  = (unsigned short*)ws;  ws += nel * 2;   // K  [B,H,N,Dh]
  unsigned short* vtw = (unsigned short*)ws;  ws += nel * 2;   // V^T[B,H,Dh,N]
  unsigned short* ow  = (unsigned short*)ws;  ws += nel * 2;   // O  [B,N,C]

  qkv_gemm_k<<<dim3(MTOT / 128, NQKV / 256), 256, 0, stream>>>(x, wqkv, qw, kw, vtw);
  attn_k<<<dim3(SEQ / 128, BATCH * NH), 256, 0, stream>>>(qw, kw, vtw, ow);
  proj_k<<<dim3(MTOT / 128, DIM / 256), 256, 0, stream>>>(ow, wp, bp, out);
}
